// HardQuadCenterRadiusTripletLoss_40767829573850
// MI455X (gfx1250) — matrix operand order repaired and checked by reading, not compiled, not run on hardware
//
#include <hip/hip_runtime.h>
#include <hip/hip_bf16.h>

// ---------------- problem constants (from reference) ----------------
constexpr int Bb = 4, Nn = 2048, Cc = 64, Hh = 512, Ww = 512;
constexpr int GS = 8, Hg = Hh / GS, Wg = Ww / GS, Mm = Hg * Wg;  // 64,64,4096
constexpr float EPSF = 1e-8f;

// ---------------- CDNA5 WMMA types ----------------
typedef __attribute__((ext_vector_type(16))) __bf16 v16bf;
typedef __attribute__((ext_vector_type(8)))  float  v8f;

struct alignas(16) u32x4 { unsigned int a, b, c, d; };
struct u32x8 { u32x4 lo, hi; };

__device__ __forceinline__ unsigned short f32_to_bf16(float f) {
  unsigned int u = __builtin_bit_cast(unsigned int, f);
  unsigned int r = u + 0x7fffu + ((u >> 16) & 1u);   // round to nearest even
  return (unsigned short)(r >> 16);
}

// A-matrix fragment (16-bit A 16x32): lane<16: e[0..7]=A[row][c0..c0+7],
// e[8..15]=A[row][c0+16..c0+23]; lanes>=16 use c0=8 (caller passes c0).
__device__ __forceinline__ v16bf load_fragA(const unsigned short* row, int c0) {
  u32x8 p;
  p.lo = *reinterpret_cast<const u32x4*>(row + c0);
  p.hi = *reinterpret_cast<const u32x4*>(row + c0 + 16);
  return __builtin_bit_cast(v16bf, p);
}

// B-matrix fragment (16-bit B 32x16): e[0..7] = cells[m][c0 .. c0+7],
// e[8..15] = cells[m][c0+16 .. c0+23]; lanes>=16 use c0=8.
__device__ __forceinline__ v16bf load_fragB(const unsigned short* row, int c0) {
  u32x8 p;
  p.lo = *reinterpret_cast<const u32x4*>(row + c0);
  p.hi = *reinterpret_cast<const u32x4*>(row + c0 + 16);
  return __builtin_bit_cast(v16bf, p);
}

// insert x into ascending sorted t[0..3], keep 4 smallest (7 min/max ops)
__device__ __forceinline__ void ins4(float (&t)[4], float x) {
  float m3 = fminf(t[3], x);
  float a = fminf(t[2], m3); float b = fmaxf(t[2], m3); t[3] = b;
  float c = fminf(t[1], a);  float d = fmaxf(t[1], a);  t[2] = d;
  float e = fminf(t[0], c);  float f = fmaxf(t[0], c);  t[1] = f; t[0] = e;
}

// ---------------- kernel 1: 8x8 average pool of desc2 -> cell means ----------------
// block = 64 threads (thread = gx), grid = B*C*Hg
__global__ void k_pool_cells(const float* __restrict__ desc2,
                             float* __restrict__ cells_f32) {
  int t  = threadIdx.x;            // gx 0..63
  int bid = blockIdx.x;
  int gy = bid & 63;
  int c  = (bid >> 6) & 63;
  int b  = bid >> 12;
  const float* base = desc2 + ((size_t)(b * Cc + c) * Hh + gy * GS) * Ww + t * GS;
  float s = 0.f;
#pragma unroll
  for (int r = 0; r < GS; r++) {
    const float4* p = reinterpret_cast<const float4*>(base + (size_t)r * Ww);
    float4 q0 = p[0], q1 = p[1];
    s += q0.x + q0.y + q0.z + q0.w + q1.x + q1.y + q1.z + q1.w;
  }
  s *= (1.0f / 64.0f);
  cells_f32[((size_t)(b * Mm + (gy * Wg + t))) * Cc + c] = s;
}

// ---------------- kernel 2: normalize cells -> bf16 rows + cn ----------------
// one wave per cell; block = 256 (8 waves), grid = B*M/8
__global__ void k_norm_cells(const float* __restrict__ cells_f32,
                             unsigned short* __restrict__ cellsB,
                             float* __restrict__ cn_ws) {
  int lane = threadIdx.x & 31;
  int wave = threadIdx.x >> 5;
  int idx  = blockIdx.x * 8 + wave;           // 0 .. B*M-1
  int b = idx >> 12, m = idx & (Mm - 1);
  const float* row = cells_f32 + (size_t)(b * Mm + m) * Cc;
  float v0 = row[lane], v1 = row[lane + 32];
  float pp = v0 * v0 + v1 * v1;
  for (int off = 16; off > 0; off >>= 1) pp += __shfl_xor(pp, off, 32);
  float L = sqrtf(pp);
  float inv = 1.0f / (L + EPSF);
  unsigned short* orow = cellsB + (size_t)(b * Mm + m) * Cc;
  orow[lane]      = f32_to_bf16(v0 * inv);
  orow[lane + 32] = f32_to_bf16(v1 * inv);
  if (lane == 0) cn_ws[b * Mm + m] = pp * inv * inv;
}

// ---------------- kernel 3: keypoint prep: bf16 A rows, an, d_pos ----------------
// one wave per keypoint; block = 256, grid = B*N/8
__global__ void k_prep_kp(const float* __restrict__ kp1_desc,
                          const float* __restrict__ wkp,
                          const float* __restrict__ desc2,
                          unsigned short* __restrict__ descB,
                          float* __restrict__ an_ws,
                          float* __restrict__ dpos_ws) {
  int lane = threadIdx.x & 31;
  int wave = threadIdx.x >> 5;
  int idx  = blockIdx.x * 8 + wave;           // 0 .. B*N-1
  int b = idx >> 11, n = idx & (Nn - 1);
  const float* arow = kp1_desc + (size_t)(b * Nn + n) * Cc;
  float a0 = arow[lane], a1 = arow[lane + 32];
  unsigned short* orow = descB + (size_t)(b * Nn + n) * Cc;
  orow[lane]      = f32_to_bf16(a0);
  orow[lane + 32] = f32_to_bf16(a1);

  float wx = wkp[(b * Nn + n) * 2 + 0];
  float wy = wkp[(b * Nn + n) * 2 + 1];
  float x = fminf(fmaxf(wx, 0.f), (float)(Ww - 1));
  float y = fminf(fmaxf(wy, 0.f), (float)(Hh - 1));
  float x0 = floorf(x), y0 = floorf(y);
  int x0i = (int)x0, y0i = (int)y0;
  int x1i = (x0i + 1 < Ww - 1) ? x0i + 1 : Ww - 1;
  int y1i = (y0i + 1 < Hh - 1) ? y0i + 1 : Hh - 1;
  float fx = x - x0, fy = y - y0;
  float w00 = (1.f - fy) * (1.f - fx), w01 = (1.f - fy) * fx;
  float w10 = fy * (1.f - fx),         w11 = fy * fx;

  float dd = 0.f, pd = 0.f, pp = 0.f;
#pragma unroll
  for (int k = 0; k < 2; k++) {
    int c = lane + k * 32;
    float a = (k == 0) ? a0 : a1;
    const float* plane = desc2 + (size_t)(b * Cc + c) * Hh * Ww;
    float v00 = plane[y0i * Ww + x0i];
    float v01 = plane[y0i * Ww + x1i];
    float v10 = plane[y1i * Ww + x0i];
    float v11 = plane[y1i * Ww + x1i];
    float pos = w00 * v00 + w01 * v01 + w10 * v10 + w11 * v11;
    dd += a * a; pd += a * pos; pp += pos * pos;
  }
  for (int off = 16; off > 0; off >>= 1) {
    dd += __shfl_xor(dd, off, 32);
    pd += __shfl_xor(pd, off, 32);
    pp += __shfl_xor(pp, off, 32);
  }
  if (lane == 0) {
    float L = sqrtf(pp), den = L + EPSF;
    float d2 = dd - 2.f * pd / den + pp / (den * den);
    dpos_ws[b * Nn + n] = sqrtf(d2 + EPSF);
    an_ws[b * Nn + n]   = dd;
  }
}

// ---------------- kernel 4: fused WMMA GEMM + mask + top-4 + hinge ----------------
// one wave per 16-keypoint tile; block = 256 (8 waves), grid = 64 (512 tiles)
__global__ void k_fused_loss(const unsigned short* __restrict__ descB,
                             const unsigned short* __restrict__ cellsB,
                             const float* __restrict__ cn_ws,
                             const float* __restrict__ an_ws,
                             const float* __restrict__ dpos_ws,
                             const float* __restrict__ wkp,
                             float* __restrict__ partials) {
  int lane = threadIdx.x & 31;
  int wave = threadIdx.x >> 5;
  int tile = blockIdx.x * 8 + wave;           // 0..511
  int b = tile >> 7;                          // /128
  int nbase = (tile & 127) * 16;
  int half = lane >> 4;                       // 0: lanes 0-15, 1: lanes 16-31
  int lm = lane & 15;                         // m column within tile

  // A fragments (K=0..31 and K=32..63) for the 16-row keypoint tile
  const unsigned short* arow = descB + (size_t)(b * Nn + nbase + lm) * Cc;
  int ca = half ? 8 : 0;
  v16bf a0 = load_fragA(arow, ca);
  v16bf a1 = load_fragA(arow, ca + 32);

  // per-lane context: 8 keypoints (VGPR rows of the C/D matrix)
  float wx[8], wy[8], anv[8], t[8][4];
#pragma unroll
  for (int v = 0; v < 8; v++) {
    int nv = nbase + half * 8 + v;
    wx[v] = wkp[(b * Nn + nv) * 2 + 0];
    wy[v] = wkp[(b * Nn + nv) * 2 + 1];
    anv[v] = an_ws[b * Nn + nv];
    t[v][0] = t[v][1] = t[v][2] = t[v][3] = 3.0e38f;
  }

  int cb = half ? 8 : 0;
  const unsigned short* cbase = cellsB + (size_t)b * Mm * Cc;
#pragma unroll 2
  for (int mt = 0; mt < Mm / 16; mt++) {
    int m = mt * 16 + lm;
    const unsigned short* brow = cbase + (size_t)m * Cc;
    // pull the next m-tile's B rows toward the WGP (global_prefetch_b8)
    __builtin_prefetch(brow + 16 * Cc, 0, 0);
    v16bf bf0 = load_fragB(brow, cb);
    v16bf bf1 = load_fragB(brow, cb + 32);
    v8f acc = {};
    acc = __builtin_amdgcn_wmma_f32_16x16x32_bf16(false, a0, false, bf0,
                                                  (short)0, acc, false, false);
    acc = __builtin_amdgcn_wmma_f32_16x16x32_bf16(false, a1, false, bf1,
                                                  (short)0, acc, false, false);
    float cnm = cn_ws[b * Mm + m];
    float cx = ((float)(m & 63) + 0.5f) * (float)GS;
    float cy = ((float)(m >> 6) + 0.5f) * (float)GS;
#pragma unroll
    for (int v = 0; v < 8; v++) {
      float s = anv[v] + cnm - 2.0f * acc[v];       // squared distance
      s = fmaxf(s, 0.0f) + EPSF;
      float dx = wx[v] - cx, dy = wy[v] - cy;
      if (dx * dx + dy * dy <= (float)(GS * GS)) s = 1e12f;  // masked -> dist 1e6
      ins4(t[v], s);
    }
  }

  // butterfly merge of per-column top-4 lists within each 16-lane half
#pragma unroll
  for (int step = 1; step < 16; step <<= 1) {
#pragma unroll
    for (int v = 0; v < 8; v++) {
      float p0 = __shfl_xor(t[v][0], step, 32);
      float p1 = __shfl_xor(t[v][1], step, 32);
      float p2 = __shfl_xor(t[v][2], step, 32);
      float p3 = __shfl_xor(t[v][3], step, 32);
      ins4(t[v], p0); ins4(t[v], p1); ins4(t[v], p2); ins4(t[v], p3);
    }
  }

  float sum = 0.f;
#pragma unroll
  for (int v = 0; v < 8; v++) {
    int nv = nbase + half * 8 + v;
    float dp = dpos_ws[b * Nn + nv];
#pragma unroll
    for (int k = 0; k < 4; k++)
      sum += fmaxf(dp - sqrtf(t[v][k]) + 1.0f, 0.0f);   // MARGIN = 1.0
  }
  float partial = (lm == 0) ? sum * (1.0f / (float)(Bb * Nn * 4)) : 0.0f;
  partial += __shfl_xor(partial, 16, 32);
  if (lane == 0) partials[tile] = partial;
}

// ---------------- kernel 5: deterministic final reduction ----------------
__global__ void k_reduce(const float* __restrict__ partials, float* __restrict__ out) {
  __shared__ float s[512];
  int t = threadIdx.x;
  s[t] = partials[t];
  __syncthreads();
  for (int off = 256; off > 0; off >>= 1) {
    if (t < off) s[t] += s[t + off];
    __syncthreads();
  }
  if (t == 0) out[0] = s[0];   // LOSS_LAMBDA = 1.0
}

// ---------------- launch ----------------
extern "C" void kernel_launch(void* const* d_in, const int* in_sizes, int n_in,
                              void* d_out, int out_size, void* d_ws, size_t ws_size,
                              hipStream_t stream) {
  const float* w_kp1    = (const float*)d_in[1];
  const float* kp1_desc = (const float*)d_in[2];
  const float* desc2    = (const float*)d_in[3];
  float* out = (float*)d_out;

  char* ws = (char*)d_ws;
  float*          cells_f32 = (float*)ws;                                   // 4 MB
  unsigned short* cellsB    = (unsigned short*)(ws + (4u << 20));           // 2 MB
  unsigned short* descB     = (unsigned short*)(ws + (6u << 20));           // 1 MB
  float*          cn_ws     = (float*)(ws + (7u << 20));                    // 64 KB
  float*          an_ws     = (float*)(ws + (7u << 20) + (64u << 10));      // 32 KB
  float*          dpos_ws   = (float*)(ws + (7u << 20) + (96u << 10));      // 32 KB
  float*          partials  = (float*)(ws + (7u << 20) + (128u << 10));     // 2 KB

  k_pool_cells<<<Bb * Cc * Hg, 64, 0, stream>>>(desc2, cells_f32);
  k_norm_cells<<<(Bb * Mm) / 8, 256, 0, stream>>>(cells_f32, cellsB, cn_ws);
  k_prep_kp<<<(Bb * Nn) / 8, 256, 0, stream>>>(kp1_desc, w_kp1, desc2,
                                               descB, an_ws, dpos_ws);
  k_fused_loss<<<(Bb * (Nn / 16)) / 8, 256, 0, stream>>>(descB, cellsB, cn_ws,
                                                         an_ws, dpos_ws, w_kp1,
                                                         partials);
  k_reduce<<<1, 512, 0, stream>>>(partials, out);
}
